// BertDecoder_63350767616740
// MI455X (gfx1250) — matrix operand order repaired and checked by reading, not compiled, not run on hardware
//
#include <hip/hip_runtime.h>
#include <hip/hip_bf16.h>
#include <math.h>

// B=1024, S=256, V=32000, E=300, H=256, ENC=768
typedef __attribute__((ext_vector_type(16))) __bf16 v16bf;
typedef __attribute__((ext_vector_type(8)))  float  v8f;
typedef __attribute__((ext_vector_type(4)))  float  f32x4;
typedef __attribute__((ext_vector_type(4)))  unsigned int u32x4;
typedef __attribute__((ext_vector_type(2)))  unsigned int u32x2;

union BF16x16 {
  unsigned short us[16];
  __bf16         bs[16];
  u32x4          q[2];
  v16bf          v;
};

// ---- fragment loaders ------------------------------------------------------
// A 16x32 bf16 (ISA 7.12.2): lane m=lane%16; chunks at K = 8*half and 16+8*half
__device__ __forceinline__ v16bf load_a_bf16(const __bf16* __restrict__ p) {
  BF16x16 u;
  u.q[0] = *(const u32x4*)(p);        // K +0..7
  u.q[1] = *(const u32x4*)(p + 16);   // K +16..23
  return u.v;
}
// B 32x16 bf16: lane n=lane%16; chunks at K = 8*half and 16+8*half
__device__ __forceinline__ v16bf load_b_bf16(const __bf16* __restrict__ p) {
  BF16x16 u;
  u.q[0] = *(const u32x4*)(p);
  u.q[1] = *(const u32x4*)(p + 16);
  return u.v;
}
// A fragment from bf16 tile staged in LDS (row stride 776 elems, padded)
__device__ __forceinline__ v16bf load_a_lds(const unsigned short* p) {
  BF16x16 u;
  u.q[0] = *(const u32x4*)(p);
  u.q[1] = *(const u32x4*)(p + 16);
  return u.v;
}

// ---- generic all-bf16 GEMM: C[M,N] = A[M,K] @ W[N,K]^T (+bias) -------------
// Block = 4 waves; each wave owns a 32(M) x 64(N) patch (2 A-frags reuse each
// B-frag twice -> 8 WMMA per 12 b128 loads).
template <bool NT_STORE>
__global__ __launch_bounds__(128) void gemm_bf16_kernel(
    const __bf16* __restrict__ A, int lda,
    const __bf16* __restrict__ W, int ldw, int wOff,
    const float* __restrict__ bias,
    float* __restrict__ C, int ldc, int N, int K)
{
  const int lane = threadIdx.x & 31;
  const int wave = threadIdx.x >> 5;
  const int hf   = lane >> 4;
  const int m16  = lane & 15;
  const int mBase = blockIdx.y * 32;
  const int nBase = blockIdx.x * 256 + wave * 64;

  v8f acc[2][4] = {};
  const __bf16* aRow0 = A + (size_t)(mBase + m16) * lda + 8 * hf;
  const __bf16* aRow1 = aRow0 + (size_t)16 * lda;
  const __bf16* wRow[4];
#pragma unroll
  for (int t = 0; t < 4; ++t)
    wRow[t] = W + (size_t)(nBase + t * 16 + m16) * ldw + wOff + 8 * hf;

  for (int k0 = 0; k0 < K; k0 += 32) {
    v16bf a0 = load_a_bf16(aRow0 + k0);
    v16bf a1 = load_a_bf16(aRow1 + k0);
#pragma unroll
    for (int t = 0; t < 4; ++t) {
      v16bf bf = load_b_bf16(wRow[t] + k0);
      acc[0][t] = __builtin_amdgcn_wmma_f32_16x16x32_bf16(
          false, a0, false, bf, (short)0, acc[0][t], false, false);
      acc[1][t] = __builtin_amdgcn_wmma_f32_16x16x32_bf16(
          false, a1, false, bf, (short)0, acc[1][t], false, false);
    }
  }
#pragma unroll
  for (int mi = 0; mi < 2; ++mi) {
#pragma unroll
    for (int t = 0; t < 4; ++t) {
      int n = nBase + t * 16 + m16;
      float bb = bias ? bias[n] : 0.0f;
#pragma unroll
      for (int r = 0; r < 8; ++r) {
        float val = acc[mi][t][r] + bb;
        float* dst = &C[(size_t)(mBase + mi * 16 + r + 8 * hf) * ldc + n];
        if (NT_STORE) __builtin_nontemporal_store(val, dst);
        else          *dst = val;
      }
    }
  }
}

// ---- fused attention scores ------------------------------------------------
// scores[b,s] = v_w . tanh(enc[b,s,:] @ w_e^T + hWh[b,:] + attn_b)
// The 16x768 fp32 enc tile is loaded+converted ONCE per block into LDS (bf16,
// row padded to 776), then all 4 waves feed WMMA from LDS.
__global__ __launch_bounds__(128) void attn_scores_kernel(
    const float*  __restrict__ enc,       // (B*S, 768) fp32, streamed NT
    const __bf16* __restrict__ attnw_bf,  // (256, 1024) bf16; w_e at col 256
    const float*  __restrict__ attn_b,    // (256)
    const float*  __restrict__ v_w,       // (256)
    const float*  __restrict__ hWh,       // (B, 256)
    float* __restrict__ scores)           // (B, S)
{
  __shared__ __align__(16) unsigned short sA[16 * 776];  // 24.25 KB bf16 tile
  __shared__ float sS[16];
  const int tid  = threadIdx.x;
  const int lane = tid & 31;
  const int wave = tid >> 5;
  const int hf   = lane >> 4;
  const int m16  = lane & 15;
  const int b  = blockIdx.x >> 4;
  const int s0 = (blockIdx.x & 15) << 4;
  const int nBase = wave * 64;

  if (tid < 16) sS[tid] = 0.0f;

  // cooperative tile fill: 16 rows x 768 cols, fp32 NT load -> bf16 LDS
  const float* base = enc + ((size_t)b * 256 + s0) * 768;
  for (int u = tid; u < 3072; u += 128) {          // 3072 float4 units
    int row = u / 192;
    int col = (u % 192) * 4;
    f32x4 v = __builtin_nontemporal_load((const f32x4*)(base + (size_t)row * 768 + col));
    union { __bf16 b4[4]; u32x2 d; } pk;
    pk.b4[0] = (__bf16)v[0]; pk.b4[1] = (__bf16)v[1];
    pk.b4[2] = (__bf16)v[2]; pk.b4[3] = (__bf16)v[3];
    *(u32x2*)&sA[row * 776 + col] = pk.d;
  }
  __syncthreads();

  v8f acc[4] = {};
  const unsigned short* aRow = &sA[m16 * 776 + 8 * hf];
  const __bf16* wRow[4];
#pragma unroll
  for (int t = 0; t < 4; ++t)
    wRow[t] = attnw_bf + (size_t)(nBase + t * 16 + m16) * 1024 + 256 + 8 * hf;

  for (int k0 = 0; k0 < 768; k0 += 32) {
    v16bf af = load_a_lds(aRow + k0);
#pragma unroll
    for (int t = 0; t < 4; ++t) {
      v16bf bf = load_b_bf16(wRow[t] + k0);
      acc[t] = __builtin_amdgcn_wmma_f32_16x16x32_bf16(
          false, af, false, bf, (short)0, acc[t], false, false);
    }
  }

  float p[8] = {0.f,0.f,0.f,0.f,0.f,0.f,0.f,0.f};
#pragma unroll
  for (int t = 0; t < 4; ++t) {
    int n = nBase + t * 16 + m16;
    float add = hWh[(size_t)b * 256 + n] + attn_b[n];
    float vw  = v_w[n];
#pragma unroll
    for (int r = 0; r < 8; ++r)
      p[r] += tanhf(acc[t][r] + add) * vw;
  }
#pragma unroll
  for (int mask = 1; mask <= 8; mask <<= 1)
#pragma unroll
    for (int r = 0; r < 8; ++r)
      p[r] += __shfl_xor(p[r], mask, 32);
  if (m16 == 0) {
#pragma unroll
    for (int r = 0; r < 8; ++r)
      atomicAdd(&sS[r + 8 * hf], p[r]);   // ds_add_f32
  }
  __syncthreads();
  if (tid < 16)
    scores[(size_t)b * 256 + s0 + tid] = sS[tid];
}

// ---- small helper kernels --------------------------------------------------
__global__ __launch_bounds__(256) void cvt_bf16_kernel(
    const float* __restrict__ src, __bf16* __restrict__ dst, int n)
{
  int i = blockIdx.x * 256 + threadIdx.x;
  if (i < n) dst[i] = (__bf16)src[i];
}

__global__ __launch_bounds__(256) void softmax_kernel(float* __restrict__ scores) {
  __shared__ float red[256];
  const int b = blockIdx.x, t = threadIdx.x;
  float v = scores[(size_t)b * 256 + t];
  red[t] = v; __syncthreads();
  for (int off = 128; off > 0; off >>= 1) {
    if (t < off) red[t] = fmaxf(red[t], red[t + off]);
    __syncthreads();
  }
  float mx = red[0]; __syncthreads();
  float e = __expf(v - mx);
  red[t] = e; __syncthreads();
  for (int off = 128; off > 0; off >>= 1) {
    if (t < off) red[t] += red[t + off];
    __syncthreads();
  }
  scores[(size_t)b * 256 + t] = e / red[0];
}

__global__ __launch_bounds__(256) void context_kernel(
    const float* __restrict__ attn, const float* __restrict__ enc,
    float* __restrict__ ctx)
{
  __shared__ float sA[256];
  const int b = blockIdx.x, t = threadIdx.x;
  sA[t] = attn[(size_t)b * 256 + t];
  __syncthreads();
  const int d = blockIdx.y * 256 + t;
  const float* e = enc + (size_t)b * 256 * 768 + d;
  float acc = 0.f;
#pragma unroll 4
  for (int s = 0; s < 256; ++s)
    acc += sA[s] * __builtin_nontemporal_load(e + (size_t)s * 768);
  ctx[(size_t)b * 768 + d] = acc;
}

// xi(bf16) = [emb[x[b]] (300) | context (768) | h (256) | zero-pad to 1344]
__global__ __launch_bounds__(256) void build_xicat_kernel(
    const int* __restrict__ x, const float* __restrict__ emb,
    const float* __restrict__ ctx, const float* __restrict__ h,
    __bf16* __restrict__ xi)
{
  const int b = blockIdx.x;
  const int j = blockIdx.y * 256 + threadIdx.x;
  if (j >= 1344) return;
  float v;
  if      (j < 300)  v = emb[(size_t)x[b] * 300 + j];
  else if (j < 1068) v = ctx[(size_t)b * 768 + (j - 300)];
  else if (j < 1324) v = h[(size_t)b * 256 + (j - 1068)];
  else               v = 0.f;
  xi[(size_t)b * 1344 + j] = (__bf16)v;
}

// Wc(bf16) = [w_ih (1068) | w_hh (256) | zero-pad], biasc = b_ih + b_hh
__global__ __launch_bounds__(256) void build_wcat_kernel(
    const float* __restrict__ w_ih, const float* __restrict__ w_hh,
    const float* __restrict__ b_ih, const float* __restrict__ b_hh,
    __bf16* __restrict__ Wc, float* __restrict__ biasc)
{
  const int n = blockIdx.x;
  const int j = blockIdx.y * 256 + threadIdx.x;
  if (j >= 1344) return;
  float v;
  if      (j < 1068) v = w_ih[(size_t)n * 1068 + j];
  else if (j < 1324) v = w_hh[(size_t)n * 256 + (j - 1068)];
  else               v = 0.f;
  Wc[(size_t)n * 1344 + j] = (__bf16)v;
  if (j == 0) biasc[n] = b_ih[n] + b_hh[n];
}

__global__ __launch_bounds__(256) void lstm_kernel(
    const float* __restrict__ gates, const float* __restrict__ c,
    float* __restrict__ h2, float* __restrict__ c2,
    __bf16* __restrict__ h2bf)
{
  const int b = blockIdx.x, j = threadIdx.x;
  const float* g = gates + (size_t)b * 1024;
  float ig = 1.f / (1.f + __expf(-g[j]));
  float fg = 1.f / (1.f + __expf(-g[j + 256]));
  float gg = tanhf(g[j + 512]);
  float og = 1.f / (1.f + __expf(-g[j + 768]));
  float cc = fg * c[(size_t)b * 256 + j] + ig * gg;
  float hh = og * tanhf(cc);
  c2[(size_t)b * 256 + j]   = cc;
  h2[(size_t)b * 256 + j]   = hh;
  h2bf[(size_t)b * 256 + j] = (__bf16)hh;
}

extern "C" void kernel_launch(void* const* d_in, const int* in_sizes, int n_in,
                              void* d_out, int out_size, void* d_ws, size_t ws_size,
                              hipStream_t stream)
{
  const int*   x      = (const int*)  d_in[0];
  const float* h      = (const float*)d_in[1];
  const float* c      = (const float*)d_in[2];
  const float* enc    = (const float*)d_in[3];
  const float* emb    = (const float*)d_in[4];
  const float* attn_w = (const float*)d_in[5];
  const float* attn_b = (const float*)d_in[6];
  const float* v_w    = (const float*)d_in[7];
  const float* w_ih   = (const float*)d_in[8];
  const float* w_hh   = (const float*)d_in[9];
  const float* b_ih   = (const float*)d_in[10];
  const float* b_hh   = (const float*)d_in[11];
  const float* out_w  = (const float*)d_in[12];
  const float* out_b  = (const float*)d_in[13];

  float* out    = (float*)d_out;
  float* logits = out;                              // 1024*32000
  float* h2     = out + (size_t)1024 * 32000;       // 1024*256
  float* c2     = h2 + 1024 * 256;                  // 1024*256

  // scratch layout (float units; all offsets 16B aligned) ~33 MB total
  float* ws       = (float*)d_ws;
  float* hWh      = ws;                         // 262144
  float* scores   = ws + 262144;                // 262144 (attn after softmax)
  float* ctx      = ws + 524288;                // 786432
  float* gates    = ws + 1310720;               // 1048576
  float* biasc    = ws + 2359296;               // 1024
  __bf16* xi_bf   = (__bf16*)(ws + 2360320);    // 1376256 bf16
  __bf16* Wc_bf   = (__bf16*)(ws + 3048448);    // 1376256 bf16
  __bf16* attnwbf = (__bf16*)(ws + 3736576);    // 262144  bf16
  __bf16* outwbf  = (__bf16*)(ws + 3867648);    // 8192000 bf16
  __bf16* h_bf    = (__bf16*)(ws + 7963648);    // 262144  bf16
  __bf16* h2_bf   = (__bf16*)(ws + 8094720);    // 262144  bf16

  // 0) one-shot weight/activation conversions to bf16
  cvt_bf16_kernel<<<dim3(1024),  256, 0, stream>>>(attn_w, attnwbf, 262144);
  cvt_bf16_kernel<<<dim3(32000), 256, 0, stream>>>(out_w,  outwbf,  8192000);
  cvt_bf16_kernel<<<dim3(1024),  256, 0, stream>>>(h,      h_bf,    262144);

  // 1) hWh = h @ w_h.T          (M=1024, N=256, K=256)
  gemm_bf16_kernel<false><<<dim3(1, 32), 128, 0, stream>>>(
      h_bf, 256, attnwbf, 1024, 0, nullptr, hWh, 256, 256, 256);
  // 2) attention scores (fused energy GEMM + tanh + v_w reduction)
  attn_scores_kernel<<<dim3(16384), 128, 0, stream>>>(
      enc, attnwbf, attn_b, v_w, hWh, scores);
  // 3) softmax over S (in place)
  softmax_kernel<<<dim3(1024), 256, 0, stream>>>(scores);
  // 4) context = attn @ enc
  context_kernel<<<dim3(1024, 3), 256, 0, stream>>>(scores, enc, ctx);
  // 5) concat inputs / weights for the LSTM gate GEMM (K padded 1324->1344)
  build_xicat_kernel<<<dim3(1024, 6), 256, 0, stream>>>(x, emb, ctx, h, xi_bf);
  build_wcat_kernel<<<dim3(1024, 6), 256, 0, stream>>>(w_ih, w_hh, b_ih, b_hh, Wc_bf, biasc);
  // 6) gates = xi @ Wc^T + biasc (M=1024, N=1024, K=1344)
  gemm_bf16_kernel<false><<<dim3(4, 32), 128, 0, stream>>>(
      xi_bf, 1344, Wc_bf, 1344, 0, biasc, gates, 1024, 1024, 1344);
  // 7) LSTM cell elementwise -> h2, c2 (into d_out) + bf16 copy of h2
  lstm_kernel<<<dim3(1024), 256, 0, stream>>>(gates, c, h2, c2, h2_bf);
  // 8) logits = h2 @ out_w^T + out_b (M=1024, N=32000, K=256), NT stores
  gemm_bf16_kernel<true><<<dim3(125, 32), 128, 0, stream>>>(
      h2_bf, 256, outwbf, 256, 0, out_b, logits, 32000, 32000, 256);
}
